// SAttention_69776038691222
// MI455X (gfx1250) — matrix operand order repaired and checked by reading, not compiled, not run on hardware
//
#include <hip/hip_runtime.h>
#include <hip/hip_bf16.h>

typedef __attribute__((ext_vector_type(16))) _Float16 v16h;
typedef __attribute__((ext_vector_type(8)))  float    v8f;

#define TOK   4096   // B*S = 2*2048
#define DMODEL 1024
#define SEQ   2048
#define NHEAD 16
#define HDIM  64

// ---------------------------------------------------------------------------
// WMMA helpers (wave32, V_WMMA_F32_16X16X32_F16)
// ---------------------------------------------------------------------------
__device__ __forceinline__ v8f wmma_f16(v16h a, v16h b, v8f c) {
    return __builtin_amdgcn_wmma_f32_16x16x32_f16(
        /*neg_a=*/false, a, /*neg_b=*/false, b,
        /*c_mod=*/(short)0, c, /*reuse_a=*/false, /*reuse_b=*/false);
}

// A-fragment: 16x32 f16, row-major source with leading dim lda.
// Lane l (0..31): row m = l&15, half hl = l>>4.
// elements 0..15 = A[m][hl*16 + i]       (K 0..15 / 16..31)
__device__ __forceinline__ v16h load_a_frag(const _Float16* A, int lda) {
    int lane = threadIdx.x & 31;
    int m  = lane & 15;
    int hl = lane >> 4;
    const _Float16* p0 = A + (size_t)m * lda + hl * 16;
    v16h a;
#pragma unroll
    for (int i = 0; i < 16; ++i) a[i] = p0[i];
    return a;
}

// B-fragment from a row-major [N x K] matrix used as B^T (i.e. Bwmma[k][n] = Bm[n][k]).
// Lane l: column n = l&15, half hl = l>>4; element i <-> K = hl*16 + i (contiguous load).
__device__ __forceinline__ v16h load_bT_frag(const _Float16* Bm, int ldb) {
    int lane = threadIdx.x & 31;
    int n  = lane & 15;
    int hl = lane >> 4;
    const _Float16* p = Bm + (size_t)n * ldb + hl * 16;
    v16h b;
#pragma unroll
    for (int i = 0; i < 16; ++i) b[i] = p[i];
    return b;
}

// B-fragment from a row-major [K x N] matrix (Bwmma[k][n] = Bm[k][n]); strided along K.
__device__ __forceinline__ v16h load_b_rm_frag(const _Float16* Bm, int ldb) {
    int lane = threadIdx.x & 31;
    int n  = lane & 15;
    int hl = lane >> 4;
    v16h b;
#pragma unroll
    for (int i = 0; i < 16; ++i) b[i] = Bm[(size_t)(hl * 16 + i) * ldb + n];
    return b;
}

// ---------------------------------------------------------------------------
// f32 -> f16 weight conversion
// ---------------------------------------------------------------------------
__global__ void __launch_bounds__(256)
cvt_f32_f16(const float* __restrict__ in, _Float16* __restrict__ out, int n) {
    int i = blockIdx.x * 256 + threadIdx.x;
    if (i < n) out[i] = (_Float16)in[i];
}

// ---------------------------------------------------------------------------
// LayerNorm over D=1024, one block (256 threads) per token.
// optional addend (residual) input; writes f32 and f16 copies of the output.
// ---------------------------------------------------------------------------
__global__ void __launch_bounds__(256)
ln_kernel(const float* __restrict__ x, const float* __restrict__ addend,
          const float* __restrict__ g, const float* __restrict__ bta,
          float* __restrict__ yf, _Float16* __restrict__ yh) {
    const int D = DMODEL;
    int t = blockIdx.x;
    const float* xr = x + (size_t)t * D;
    const float* ar = addend ? addend + (size_t)t * D : nullptr;

    float v[4];
    float s = 0.f;
#pragma unroll
    for (int j = 0; j < 4; ++j) {
        int c = threadIdx.x + j * 256;
        v[j] = xr[c] + (ar ? ar[c] : 0.f);
        s += v[j];
    }
    __shared__ float red[8];
#pragma unroll
    for (int i = 16; i >= 1; i >>= 1) s += __shfl_xor(s, i);
    if ((threadIdx.x & 31) == 0) red[threadIdx.x >> 5] = s;
    __syncthreads();
    float mean = 0.f;
#pragma unroll
    for (int w = 0; w < 8; ++w) mean += red[w];
    mean *= (1.f / D);
    __syncthreads();

    float s2 = 0.f;
#pragma unroll
    for (int j = 0; j < 4; ++j) { float d = v[j] - mean; s2 += d * d; }
#pragma unroll
    for (int i = 16; i >= 1; i >>= 1) s2 += __shfl_xor(s2, i);
    if ((threadIdx.x & 31) == 0) red[threadIdx.x >> 5] = s2;
    __syncthreads();
    float var = 0.f;
#pragma unroll
    for (int w = 0; w < 8; ++w) var += red[w];
    var *= (1.f / D);
    float rstd = rsqrtf(var + 1e-5f);

#pragma unroll
    for (int j = 0; j < 4; ++j) {
        int c = threadIdx.x + j * 256;
        float o = (v[j] - mean) * rstd * g[c] + bta[c];
        if (yf) yf[(size_t)t * D + c] = o;
        yh[(size_t)t * D + c] = (_Float16)o;
    }
}

// ---------------------------------------------------------------------------
// Generic WMMA GEMM: C[M x N] = A[M x K] * Bm[N x K]^T  (+bias, +ReLU, +residual)
// Block: 256 threads = 8 waves arranged 4(M) x 2(N); block tile 128x128.
// Wave tile 32x64: 8 accumulators, 8 WMMAs per K-step of 32.
// Branch-free ping-pong double buffering: each half-step's fragments are
// loaded a full WMMA group ahead in a disjoint register buffer, so there are
// no buffer-rotation copies and the waits cover loads issued one step early.
// Requires K to be a multiple of 64 (K = 1024 for every call here).
// ---------------------------------------------------------------------------
__global__ void __launch_bounds__(256)
gemm_wmma(const _Float16* __restrict__ A, const _Float16* __restrict__ Bm,
          const float* __restrict__ bias, const float* __restrict__ res,
          float* __restrict__ outf, _Float16* __restrict__ outh,
          int M, int N, int K, int relu) {
    int wave = threadIdx.x >> 5;
    int mw = wave & 3;      // 4 waves along M
    int nw = wave >> 2;     // 2 waves along N
    int row0 = blockIdx.x * 128 + mw * 32;
    int col0 = blockIdx.y * 128 + nw * 64;
    int lane = threadIdx.x & 31;
    int n  = lane & 15;
    int hl = lane >> 4;

    const _Float16* Abase = A  + (size_t)row0 * K;
    const _Float16* Bbase = Bm + (size_t)col0 * K;

    v8f zero = {};
    v8f acc[2][4];
#pragma unroll
    for (int s = 0; s < 2; ++s)
#pragma unroll
        for (int j = 0; j < 4; ++j) acc[s][j] = zero;

    v16h a0[2], b0[4], a1[2], b1[4];

    auto loadstep = [&](int k0, v16h* aa, v16h* bb) {
        aa[0] = load_a_frag(Abase + k0, K);
        aa[1] = load_a_frag(Abase + (size_t)16 * K + k0, K);
#pragma unroll
        for (int j = 0; j < 4; ++j)
            bb[j] = load_bT_frag(Bbase + (size_t)(j * 16) * K + k0, K);
    };
    auto compute = [&](v16h* aa, v16h* bb) {
#pragma unroll
        for (int s = 0; s < 2; ++s)
#pragma unroll
            for (int j = 0; j < 4; ++j)
                acc[s][j] = wmma_f16(aa[s], bb[j], acc[s][j]);
    };

    const int nsteps = K >> 5;           // K/32, even
    loadstep(0, a0, b0);
    for (int s = 0; s + 2 < nsteps; s += 2) {
        loadstep((s + 1) * 32, a1, b1);
        compute(a0, b0);
        loadstep((s + 2) * 32, a0, b0);
        compute(a1, b1);
    }
    loadstep((nsteps - 1) * 32, a1, b1);
    compute(a0, b0);
    compute(a1, b1);

#pragma unroll
    for (int s = 0; s < 2; ++s) {
#pragma unroll
        for (int j = 0; j < 4; ++j) {
#pragma unroll
            for (int r = 0; r < 8; ++r) {
                int row = row0 + s * 16 + hl * 8 + r;
                int col = col0 + j * 16 + n;
                float vv = acc[s][j][r];
                if (bias) vv += bias[col];
                if (relu) vv = fmaxf(vv, 0.f);
                if (res)  vv += res[(size_t)row * N + col];
                if (outf) outf[(size_t)row * N + col] = vv;
                if (outh) outh[(size_t)row * N + col] = (_Float16)vv;
            }
        }
    }
}

// ---------------------------------------------------------------------------
// Flash attention: one wave per (batch, head, 16-query tile). S=2048, HDIM=64.
// q,k,v are f16 [TOK x 1024] ([B,S,H,HD]); att output f32 same layout.
// Scores via WMMA (K-dim 64 = 2 chained 16x16x32), online softmax, exp(P)
// staged through per-wave LDS (DS in-order per wave) to A-fragment layout,
// PV accumulation via WMMA with running rescale. temp = sqrt(64) = 8.
// Next key-tile K fragments and current V fragments are loaded before the
// softmax VALU section so their latency hides behind exp/shfl work.
// ---------------------------------------------------------------------------
__global__ void __launch_bounds__(256)
attn_kernel(const _Float16* __restrict__ q, const _Float16* __restrict__ k,
            const _Float16* __restrict__ v, float* __restrict__ att) {
    const int S = SEQ, H = NHEAD, HD = HDIM, LD = DMODEL;
    __shared__ _Float16 pbuf[8][16 * 32];

    int wave = threadIdx.x >> 5;
    int w = blockIdx.x * 8 + wave;
    const int NQT = S / 16;  // 128
    int qt   = w % NQT;
    int head = (w / NQT) % H;
    int bb   = w / (NQT * H);

    int lane = threadIdx.x & 31;
    int n  = lane & 15;
    int hl = lane >> 4;

    const _Float16* Q  = q + (size_t)(bb * S + qt * 16) * LD + head * HD;
    const _Float16* Kb = k + (size_t)(bb * S) * LD + head * HD;
    const _Float16* Vb = v + (size_t)(bb * S) * LD + head * HD;

    v16h aq0 = load_a_frag(Q, LD);       // d = 0..31
    v16h aq1 = load_a_frag(Q + 32, LD);  // d = 32..63

    v8f zero = {};
    v8f o[4] = {zero, zero, zero, zero};
    float mrow[8], lrow[8];
#pragma unroll
    for (int r = 0; r < 8; ++r) { mrow[r] = -1e30f; lrow[r] = 0.f; }

    _Float16* pw = &pbuf[wave][0];

    // prologue: key fragments for kk = 0
    v16h kb[4];
    kb[0] = load_bT_frag(Kb, LD);
    kb[1] = load_bT_frag(Kb + 32, LD);
    kb[2] = load_bT_frag(Kb + (size_t)16 * LD, LD);
    kb[3] = load_bT_frag(Kb + (size_t)16 * LD + 32, LD);

    for (int kk = 0; kk < S; kk += 32) {
        // scores for key tiles [kk, kk+16) and [kk+16, kk+32)
        v8f s0 = zero, s1 = zero;
        s0 = wmma_f16(aq0, kb[0], s0);
        s0 = wmma_f16(aq1, kb[1], s0);
        s1 = wmma_f16(aq0, kb[2], s1);
        s1 = wmma_f16(aq1, kb[3], s1);

        // prefetch next key tile + this tile's V fragments (hide behind softmax)
        int kn = kk + 32;
        if (kn < S) {
            kb[0] = load_bT_frag(Kb + (size_t)kn * LD, LD);
            kb[1] = load_bT_frag(Kb + (size_t)kn * LD + 32, LD);
            kb[2] = load_bT_frag(Kb + (size_t)(kn + 16) * LD, LD);
            kb[3] = load_bT_frag(Kb + (size_t)(kn + 16) * LD + 32, LD);
        }
        v16h bv[4];
#pragma unroll
        for (int j = 0; j < 4; ++j)
            bv[j] = load_b_rm_frag(Vb + (size_t)kk * LD + j * 16, LD);

#pragma unroll
        for (int r = 0; r < 8; ++r) {
            float a0 = s0[r] * 0.125f;   // 1/sqrt(HDIM)
            float a1 = s1[r] * 0.125f;
            float tmax = fmaxf(a0, a1);
#pragma unroll
            for (int i = 8; i >= 1; i >>= 1) tmax = fmaxf(tmax, __shfl_xor(tmax, i, 16));
            float mnew = fmaxf(mrow[r], tmax);
            float corr = __expf(mrow[r] - mnew);
            float p0 = __expf(a0 - mnew);
            float p1 = __expf(a1 - mnew);
            float ps = p0 + p1;
#pragma unroll
            for (int i = 8; i >= 1; i >>= 1) ps += __shfl_xor(ps, i, 16);
            lrow[r] = lrow[r] * corr + ps;
            mrow[r] = mnew;
#pragma unroll
            for (int j = 0; j < 4; ++j) o[j][r] *= corr;
            // stash probabilities row-major 16x32 (row = hl*8+r, col = n / 16+n)
            pw[(hl * 8 + r) * 32 + n]      = (_Float16)p0;
            pw[(hl * 8 + r) * 32 + 16 + n] = (_Float16)p1;
        }

        // reload as A-fragment (same wave; DS ops are in-order per wave)
        v16h ap = load_a_frag(pw, 32);
#pragma unroll
        for (int j = 0; j < 4; ++j)
            o[j] = wmma_f16(ap, bv[j], o[j]);
    }

#pragma unroll
    for (int r = 0; r < 8; ++r) {
        float inv = 1.f / lrow[r];
        int row = bb * S + qt * 16 + hl * 8 + r;
#pragma unroll
        for (int j = 0; j < 4; ++j)
            att[(size_t)row * LD + head * HD + j * 16 + n] = o[j][r] * inv;
    }
}

// ---------------------------------------------------------------------------
// Host-side launch
// ---------------------------------------------------------------------------
extern "C" void kernel_launch(void* const* d_in, const int* in_sizes, int n_in,
                              void* d_out, int out_size, void* d_ws, size_t ws_size,
                              hipStream_t stream) {
    const float* x   = (const float*)d_in[0];
    const float* Wq  = (const float*)d_in[1];
    const float* Wk  = (const float*)d_in[2];
    const float* Wv  = (const float*)d_in[3];
    const float* g1  = (const float*)d_in[4];
    const float* b1  = (const float*)d_in[5];
    const float* g2  = (const float*)d_in[6];
    const float* b2  = (const float*)d_in[7];
    const float* W1  = (const float*)d_in[8];
    const float* bf1 = (const float*)d_in[9];
    const float* W2  = (const float*)d_in[10];
    const float* bf2 = (const float*)d_in[11];
    float* out = (float*)d_out;

    const size_t NTD = (size_t)TOK * DMODEL;       // 4M elements
    const size_t WSZ = (size_t)DMODEL * DMODEL;    // 1M elements

    // workspace carve-up
    char* p = (char*)d_ws;
    float*    xn   = (float*)p;            p += NTD * sizeof(float);
    _Float16* xnh  = (_Float16*)p;         p += NTD * sizeof(_Float16);
    _Float16* wqh  = (_Float16*)p;         p += WSZ * sizeof(_Float16);
    _Float16* wkh  = (_Float16*)p;         p += WSZ * sizeof(_Float16);
    _Float16* wvh  = (_Float16*)p;         p += WSZ * sizeof(_Float16);
    _Float16* w1h  = (_Float16*)p;         p += WSZ * sizeof(_Float16);
    _Float16* w2h  = (_Float16*)p;         p += WSZ * sizeof(_Float16);
    _Float16* qh   = (_Float16*)p;         p += NTD * sizeof(_Float16);
    _Float16* kh   = (_Float16*)p;         p += NTD * sizeof(_Float16);
    _Float16* vh   = (_Float16*)p;         p += NTD * sizeof(_Float16);
    float*    attf = (float*)p;            p += NTD * sizeof(float);
    float*    xt   = (float*)p;            p += NTD * sizeof(float);
    _Float16* xth  = (_Float16*)p;         p += NTD * sizeof(_Float16);
    _Float16* hmid = (_Float16*)p;         p += NTD * sizeof(_Float16);
    (void)ws_size; (void)in_sizes; (void)n_in; (void)out_size;

    // 1) weights to f16
    {
        int blocks = (int)((WSZ + 255) / 256);
        cvt_f32_f16<<<blocks, 256, 0, stream>>>(Wq, wqh, (int)WSZ);
        cvt_f32_f16<<<blocks, 256, 0, stream>>>(Wk, wkh, (int)WSZ);
        cvt_f32_f16<<<blocks, 256, 0, stream>>>(Wv, wvh, (int)WSZ);
        cvt_f32_f16<<<blocks, 256, 0, stream>>>(W1, w1h, (int)WSZ);
        cvt_f32_f16<<<blocks, 256, 0, stream>>>(W2, w2h, (int)WSZ);
    }

    // 2) LN1: xn = LN(x)
    ln_kernel<<<TOK, 256, 0, stream>>>(x, nullptr, g1, b1, xn, xnh);

    // 3) Q/K/V projections: q = xn @ Wq^T  (f16 outputs)
    dim3 ggrid(TOK / 128, DMODEL / 128);
    gemm_wmma<<<ggrid, 256, 0, stream>>>(xnh, wqh, nullptr, nullptr, nullptr, qh,
                                         TOK, DMODEL, DMODEL, 0);
    gemm_wmma<<<ggrid, 256, 0, stream>>>(xnh, wkh, nullptr, nullptr, nullptr, kh,
                                         TOK, DMODEL, DMODEL, 0);
    gemm_wmma<<<ggrid, 256, 0, stream>>>(xnh, wvh, nullptr, nullptr, nullptr, vh,
                                         TOK, DMODEL, DMODEL, 0);

    // 4) flash attention: 2*16*128 = 4096 wave-tiles, 8 waves/block
    attn_kernel<<<4096 / 8, 256, 0, stream>>>(qh, kh, vh, attf);

    // 5) LN2 with residual: xt = LN(xn + att)
    ln_kernel<<<TOK, 256, 0, stream>>>(xn, attf, g2, b2, xt, xth);

    // 6) FFN1: h = relu(xt @ W1^T + bf1)  (f16 out)
    gemm_wmma<<<ggrid, 256, 0, stream>>>(xth, w1h, bf1, nullptr, nullptr, hmid,
                                         TOK, DMODEL, DMODEL, 1);

    // 7) FFN2 + residual: out = xt + (h @ W2^T + bf2)  (f32 out)
    gemm_wmma<<<ggrid, 256, 0, stream>>>(hmid, w2h, bf2, xt, out, nullptr,
                                         TOK, DMODEL, DMODEL, 0);
}
